// adapter_35038343201248
// MI455X (gfx1250) — 8-bit matrix operand order repaired and checked by reading, compiled, not run on hardware
//
#include <hip/hip_runtime.h>

typedef int   v4i __attribute__((ext_vector_type(4)));
typedef int   v8i __attribute__((ext_vector_type(8)));
typedef float v4f __attribute__((ext_vector_type(4)));

#define HW        3136
#define CIN       256
#define CMID      768
#define NT        128
#define XB_STRIDE 288
#define S1_STRIDE 800

__device__ __forceinline__ int swz_off(int c) {
  const int chunk = (c >> 4) & 3;
  const int perm  = ((chunk & 1) << 1) | (chunk >> 1);
  return (c & ~63) | (perm << 4) | (c & 15);
}

__device__ __forceinline__ unsigned pack_sign4(float a, float b, float c, float d) {
  const unsigned f0 = __float_as_uint(a), f1 = __float_as_uint(b);
  const unsigned f2 = __float_as_uint(c), f3 = __float_as_uint(d);
  const unsigned p01 = __builtin_amdgcn_perm(f1, f0, 0x0C0C0703u);
  const unsigned p23 = __builtin_amdgcn_perm(f3, f2, 0x07030C0Cu);
  unsigned m = (p01 | p23) & 0x80808080u;
  m |= m >> 1; m |= m >> 2; m |= m >> 4;
  return m | 0x01010101u;
}

__global__ __launch_bounds__(256) void bnn_prepack(
    const float* __restrict__ w1, const float* __restrict__ w2,
    const float* __restrict__ g1, const float* __restrict__ b1,
    const float* __restrict__ m1, const float* __restrict__ v1,
    const float* __restrict__ g2, const float* __restrict__ b2,
    const float* __restrict__ m2, const float* __restrict__ v2,
    char* __restrict__ w1f, char* __restrict__ w2f,
    int* __restrict__ thr1, float* __restrict__ inv2, float* __restrict__ add2)
{
  const int idx = blockIdx.x * 256 + threadIdx.x;
  if (idx < 12288) {
    const bool is1 = idx < 6144;
    const int  fid = (is1 ? idx : (idx - 6144)) >> 5;
    const int  lane = idx & 31;
    const int  mt = is1 ? (fid >> 2) : (fid / 12);
    const int  kt = is1 ? (fid & 3)  : (fid % 12);
    const int  ml = lane & 15, khalf = lane >> 4;
    const int  row = mt * 16 + ml;
    const int  K   = is1 ? CIN : CMID;
    const float* w = is1 ? w1 : w2;
    unsigned wrd[8];
#pragma unroll
    for (int v = 0; v < 8; ++v) {
      unsigned u = 0;
#pragma unroll
      for (int bb = 0; bb < 4; ++bb) {
        const int kl = (((v >> 2) << 1) | khalf) * 16 + (v & 3) * 4 + bb;
        const float wv = w[(size_t)row * K + kt * 64 + kl];
        u |= ((wv >= 0.f) ? 0x01u : 0xFFu) << (8 * bb);
      }
      wrd[v] = u;
    }
    unsigned* dst = (unsigned*)((is1 ? w1f : w2f) + (size_t)fid * 1024 + lane * 32);
#pragma unroll
    for (int v = 0; v < 8; ++v) dst[v] = wrd[v];
  } else if (idx < 13056) {
    const int o = idx - 12288;
    const float iv = g1[o] * rsqrtf(v1[o] + 1e-5f);
    thr1[o] = (int)ceilf(m1[o] - b1[o] / iv);
  } else if (idx < 13312) {
    const int o = idx - 13056;
    const float iv = g2[o] * rsqrtf(v2[o] + 1e-5f);
    inv2[o] = iv;
    add2[o] = b2[o] - m2[o] * iv;
  }
}

__global__ __launch_bounds__(256) void bnn_fused(
    const float* __restrict__ x,
    const char*  __restrict__ w1f, const char* __restrict__ w2f,
    const int*   __restrict__ thr1,
    const float* __restrict__ inv2, const float* __restrict__ add2,
    float* __restrict__ out)
{
  __shared__ __align__(32) char xb[NT * XB_STRIDE];
  __shared__ __align__(32) char s1[NT * S1_STRIDE];

  const int t = threadIdx.x;
  const int pbase = blockIdx.x * NT;

  {
    const int p  = t & (NT - 1);
    const int cb = (t >> 7) * 128;
    const int pg = pbase + p;
    const int b  = pg / HW;
    const int s  = pg % HW;
    const float* xp = x + ((size_t)b * CIN) * HW + s;
#pragma unroll 4
    for (int c4 = 0; c4 < 32; ++c4) {
      const int c = cb + c4 * 4;
      const unsigned u = pack_sign4(xp[(size_t)(c + 0) * HW],
                                    xp[(size_t)(c + 1) * HW],
                                    xp[(size_t)(c + 2) * HW],
                                    xp[(size_t)(c + 3) * HW]);
      *(unsigned*)&xb[p * XB_STRIDE + swz_off(c)] = u;
    }
  }
  __syncthreads();

  const int wv    = t >> 5;
  const int lane  = t & 31;
  const int nloc  = lane & 15;
  const int khalf = lane >> 4;
  const int pix   = wv * 16 + nloc;

  v8i bfrag[4];
#pragma unroll
  for (int kt = 0; kt < 4; ++kt)
    bfrag[kt] = *(const v8i*)&xb[pix * XB_STRIDE + kt * 64 + khalf * 32];

  for (int mt = 0; mt < 48; ++mt) {
    v8i acc = {0, 0, 0, 0, 0, 0, 0, 0};
    const char* wb = w1f + (size_t)(mt * 4) * 1024 + lane * 32;
#pragma unroll
    for (int kt = 0; kt < 4; ++kt) {
      v8i a = *(const v8i*)(wb + kt * 1024);
      acc = __builtin_amdgcn_wmma_i32_16x16x64_iu8(
           true, a,  true, bfrag[kt], acc, false, false);
    }
    const int ob = mt * 16 + khalf * 8;
    const v4i th0 = *(const v4i*)&thr1[ob];
    const v4i th1 = *(const v4i*)&thr1[ob + 4];
    unsigned lo = 0, hi = 0;
#pragma unroll
    for (int r = 0; r < 4; ++r) {
      lo |= ((acc[r]     >= th0[r]) ? 0x01u : 0xFFu) << (8 * r);
      hi |= ((acc[r + 4] >= th1[r]) ? 0x01u : 0xFFu) << (8 * r);
    }
    unsigned* dp = (unsigned*)&s1[pix * S1_STRIDE + swz_off(ob)];
    dp[0] = lo; dp[1] = hi;
  }

  v8i bfrag2[12];
#pragma unroll
  for (int kt = 0; kt < 12; ++kt)
    bfrag2[kt] = *(const v8i*)&s1[pix * S1_STRIDE + kt * 64 + khalf * 32];

  const int pg = pbase + pix;
  const int b  = pg / HW;
  const int s  = pg % HW;
  for (int mt = 0; mt < 16; ++mt) {
    v8i acc = {0, 0, 0, 0, 0, 0, 0, 0};
    const char* wb = w2f + (size_t)(mt * 12) * 1024 + lane * 32;
#pragma unroll
    for (int kt = 0; kt < 12; ++kt) {
      v8i a = *(const v8i*)(wb + kt * 1024);
      acc = __builtin_amdgcn_wmma_i32_16x16x64_iu8(
          true, a, true, bfrag2[kt], acc, false, false);
    }
    const int cb = mt * 16 + khalf * 8;
    const v4f iv0 = *(const v4f*)&inv2[cb];
    const v4f iv1 = *(const v4f*)&inv2[cb + 4];
    const v4f ad0 = *(const v4f*)&add2[cb];
    const v4f ad1 = *(const v4f*)&add2[cb + 4];
    const size_t base = ((size_t)b * CIN + cb) * HW + s;
#pragma unroll
    for (int r = 0; r < 8; ++r) {
      const float ivv = (r < 4) ? iv0[r] : iv1[r - 4];
      const float adv = (r < 4) ? ad0[r] : ad1[r - 4];
      const size_t gi = base + (size_t)r * HW;
      out[gi] = (float)acc[r] * ivv + adv + x[gi];
    }
  }
}

extern "C" void kernel_launch(void* const* d_in, const int* in_sizes, int n_in,
                              void* d_out, int out_size, void* d_ws, size_t ws_size,
                              hipStream_t stream) {
  const float* x  = (const float*)d_in[0];
  const float* w1 = (const float*)d_in[1];
  const float* w2 = (const float*)d_in[2];
  const float* g1 = (const float*)d_in[3];
  const float* b1 = (const float*)d_in[4];
  const float* m1 = (const float*)d_in[5];
  const float* v1 = (const float*)d_in[6];
  const float* g2 = (const float*)d_in[7];
  const float* b2 = (const float*)d_in[8];
  const float* m2 = (const float*)d_in[9];
  const float* v2 = (const float*)d_in[10];

  char* ws = (char*)d_ws;
  char*  w1f  = ws;
  char*  w2f  = ws + 196608;
  int*   thr1 = (int*)  (ws + 393216);
  float* inv2 = (float*)(ws + 396288);
  float* add2 = (float*)(ws + 397312);

  hipLaunchKernelGGL(bnn_prepack, dim3(52), dim3(256), 0, stream,
                     w1, w2, g1, b1, m1, v1, g2, b2, m2, v2,
                     w1f, w2f, thr1, inv2, add2);

  hipLaunchKernelGGL(bnn_fused, dim3(784), dim3(256), 0, stream,
                     x, w1f, w2f, thr1, inv2, add2, (float*)d_out);
}
